// RestaurantRecommenderGNN_44220983280336
// MI455X (gfx1250) — matrix operand order repaired and checked by reading, not compiled, not run on hardware
//
#include <hip/hip_runtime.h>
#include <hip/hip_bf16.h>
#include <cstdint>
#include <cstddef>

// ---------------- problem constants ----------------
#define NU 100000
#define NR 50000
#define NC 500
#define E_REV 2000000
#define E_BEL 150000

typedef __bf16 bf16;
typedef bf16  v16bf __attribute__((ext_vector_type(16)));
typedef float v8f   __attribute__((ext_vector_type(8)));

// =====================================================================
// GEMM: C[M,64] (+)= A[M,K] @ W[K,64]   (K in {32,64}, bf16 WMMA, f32 acc)
// block = 128 threads = 4 waves; wave w owns cols [16w,16w+16); 32-row tile
// Tiles are staged into LDS pre-swizzled into WMMA fragment order so each
// lane fetches its whole fragment as one contiguous 32B (v16bf) read.
// =====================================================================
template <int K, bool ACC>
__launch_bounds__(128)
__global__ void gemm_wmma_kernel(const float* __restrict__ A,
                                 const float* __restrict__ W,
                                 float* __restrict__ C, int M) {
  constexpr int KS = K / 32;                 // k-steps
  __shared__ __align__(32) bf16 sAf[2][KS][32][16];   // [rowtile][step][lane][i]
  __shared__ __align__(32) bf16 sBf[4][KS][32][16];   // [ntile][step][lane][i]
  const int tid  = threadIdx.x;
  const int row0 = blockIdx.x * 32;

  // ---- stage A (32 x K) in fragment order ----
  // A frag mapping: lane = m + 16*half ; kloc = 2j + (j>=4?8:0) + 8*half + p ; i = 2j+p
  for (int idx = tid; idx < 32 * K; idx += 128) {
    int r = idx / K, k = idx - r * K;
    int row  = row0 + r;
    float v  = (row < M) ? A[(size_t)row * K + k] : 0.0f;
    int step = k >> 5, kl = k & 31;
    int half = (kl >> 3) & 1;
    int j    = ((kl & 16) ? 4 : 0) + ((kl & 7) >> 1);
    int i    = 2 * j + (kl & 1);
    sAf[r >> 4][step][(r & 15) + 16 * half][i] = (bf16)v;
  }
  // ---- stage W (K x 64) in fragment order ----
  // B frag mapping: lane = (n&15) + 16*half ; kloc: half=(kl>>3)&1, i = (kl&7) + (kl&16 ? 8 : 0)
  for (int idx = tid; idx < K * 64; idx += 128) {
    int k = idx >> 6, n = idx & 63;
    int step = k >> 5, kl = k & 31;
    int half = (kl >> 3) & 1;
    sBf[n >> 4][step][(n & 15) + 16 * half][(kl & 7) + ((kl & 16) ? 8 : 0)] = (bf16)W[idx];
  }
  __syncthreads();

  const int wave = tid >> 5;
  const int lane = tid & 31;
  const int half = lane >> 4;
  const int n    = lane & 15;

  v8f acc0 = {}, acc1 = {};
#pragma unroll
  for (int s = 0; s < KS; ++s) {
    v16bf b  = *(const v16bf*)&sBf[wave][s][lane][0];
    v16bf a0 = *(const v16bf*)&sAf[0][s][lane][0];
    v16bf a1 = *(const v16bf*)&sAf[1][s][lane][0];
    acc0 = __builtin_amdgcn_wmma_f32_16x16x32_bf16(false, a0, false, b,
                                                   (short)0, acc0, false, false);
    acc1 = __builtin_amdgcn_wmma_f32_16x16x32_bf16(false, a1, false, b,
                                                   (short)0, acc1, false, false);
  }

  // ---- epilogue: C layout lane->col = n, VGPR i -> row i + 8*half ----
  const int col = wave * 16 + n;
  if (row0 + 32 <= M) {                       // fast path, no guards
#pragma unroll
    for (int i = 0; i < 8; ++i) {
      size_t o0 = (size_t)(row0 + i + 8 * half) * 64 + col;
      size_t o1 = o0 + (size_t)16 * 64;
      float v0 = acc0[i], v1 = acc1[i];
      if (ACC) { v0 += C[o0]; v1 += C[o1]; }
      C[o0] = v0; C[o1] = v1;
    }
  } else {
#pragma unroll
    for (int i = 0; i < 8; ++i) {
      int r0 = row0 + i + 8 * half, r1 = r0 + 16;
      if (r0 < M) {
        size_t o = (size_t)r0 * 64 + col;
        float v = acc0[i]; if (ACC) v += C[o]; C[o] = v;
      }
      if (r1 < M) {
        size_t o = (size_t)r1 * 64 + col;
        float v = acc1[i]; if (ACC) v += C[o]; C[o] = v;
      }
    }
  }
}

// ---------------- small helpers ----------------
__global__ void fill_f32_kernel(float* p, float v, long n) {
  long i = (long)blockIdx.x * blockDim.x + threadIdx.x;
  if (i < n) p[i] = v;
}
__global__ void fill_u32_kernel(unsigned* p, unsigned v, long n) {
  long i = (long)blockIdx.x * blockDim.x + threadIdx.x;
  if (i < n) p[i] = v;
}

// alpha[row] = dot(H[row,0:64], att[0:64]); one wave32 per row
__global__ void rowdot_kernel(const float* __restrict__ H,
                              const float* __restrict__ att,
                              float* __restrict__ out, int M) {
  int wid = threadIdx.x >> 5, lane = threadIdx.x & 31;
  int row = blockIdx.x * 4 + wid;
  if (row >= M) return;
  const float* h = H + (size_t)row * 64;
  float v = h[lane] * att[lane] + h[lane + 32] * att[lane + 32];
  for (int o = 16; o > 0; o >>= 1) v += __shfl_xor(v, o, 32);
  if (lane == 0) out[row] = v;
}

// we_att[k] = sum_c We[k,c]*att_e[c]  (fold (ea@We)@att_e -> ea@we_att)
__global__ void weatt_kernel(const float* __restrict__ We,
                             const float* __restrict__ atte,
                             float* __restrict__ out) {
  int k = threadIdx.x;
  if (k < 16) {
    float s = 0.f;
    for (int c = 0; c < 64; ++c) s += We[k * 64 + c] * atte[c];
    out[k] = s;
  }
}

// scatter edge attrs: esum[dst] += ea[e], ecnt[dst] += 1
__global__ void ea_scatter_kernel(const int* __restrict__ dst,
                                  const float* __restrict__ ea,
                                  float* __restrict__ esum,
                                  float* __restrict__ ecnt, int E) {
  long idx = (long)blockIdx.x * blockDim.x + threadIdx.x;
  if (idx >= (long)E * 16) return;
  int e = (int)(idx >> 4), c = (int)(idx & 15);
  int d = dst[e];
  atomicAdd(&esum[(size_t)d * 16 + c], ea[idx]);
  if (c == 0) atomicAdd(&ecnt[d], 1.0f);
}

// order-preserving float<->uint key for atomic float max
__device__ __forceinline__ unsigned fkey(float f) {
  unsigned u = __float_as_uint(f);
  return (u & 0x80000000u) ? ~u : (u | 0x80000000u);
}
__device__ __forceinline__ float fkey_dec(unsigned k) {
  return __uint_as_float((k & 0x80000000u) ? (k & 0x7fffffffu) : ~k);
}

// GAT pass1: per-edge logit (leaky relu) + segment max over dst
__global__ void gat_pass1_kernel(const int* __restrict__ src,
                                 const int* __restrict__ dst,
                                 const float* __restrict__ ea,
                                 const float* __restrict__ esum,
                                 const float* __restrict__ ecnt,
                                 const float* __restrict__ alpha_s,
                                 const float* __restrict__ alpha_d,
                                 const float* __restrict__ we_att,
                                 float* __restrict__ logits,
                                 unsigned* __restrict__ mkey,
                                 int E, int n_loop) {
  int e = blockIdx.x * blockDim.x + threadIdx.x;
  if (e >= E + n_loop) return;
  int s, d; float edot = 0.f;
  if (e < E) {
    s = src[e]; d = dst[e];
    const float* ep = ea + (size_t)e * 16;
#pragma unroll
    for (int c = 0; c < 16; ++c) edot += ep[c] * we_att[c];
  } else {                       // self loop: attr = mean of incoming attrs
    int i = e - E; s = i; d = i;
    float inv = 1.0f / fmaxf(ecnt[i], 1.0f);
    const float* ep = esum + (size_t)i * 16;
#pragma unroll
    for (int c = 0; c < 16; ++c) edot += ep[c] * inv * we_att[c];
  }
  float a = alpha_s[s] + alpha_d[d] + edot;
  a = (a > 0.f) ? a : 0.2f * a;  // leaky_relu(0.2)
  logits[e] = a;
  atomicMax(&mkey[d], fkey(a));
}

// GAT pass2: e = exp(a - max[dst]); segment sum
__global__ void gat_pass2_kernel(const int* __restrict__ dst,
                                 float* __restrict__ logits,
                                 const unsigned* __restrict__ mkey,
                                 float* __restrict__ ssum, int E, int n_loop) {
  int e = blockIdx.x * blockDim.x + threadIdx.x;
  if (e >= E + n_loop) return;
  int d = (e < E) ? dst[e] : (e - E);
  float ex = __expf(logits[e] - fkey_dec(mkey[d]));
  logits[e] = ex;
  atomicAdd(&ssum[d], ex);
}

// GAT pass3: out[dst] += (e/sum) * hs[src]  (channel axis on lanes)
__global__ void gat_pass3_kernel(const int* __restrict__ src,
                                 const int* __restrict__ dst,
                                 const float* __restrict__ logits,
                                 const float* __restrict__ ssum,
                                 const float* __restrict__ hs,
                                 float* __restrict__ out, int E, int n_loop) {
  long idx = (long)blockIdx.x * blockDim.x + threadIdx.x;
  if (idx >= (long)(E + n_loop) * 64) return;
  int e = (int)(idx >> 6), c = (int)(idx & 63);
  int s = (e < E) ? src[e] : (e - E);
  int d = (e < E) ? dst[e] : (e - E);
  float w = logits[e] / (ssum[d] + 1e-16f);
  atomicAdd(&out[(size_t)d * 64 + c], w * hs[(size_t)s * 64 + c]);
}

// SAGE: agg[dst] += x[src]; cnt[dst] += 1
__global__ void sage_scatter_kernel(const int* __restrict__ src,
                                    const int* __restrict__ dst,
                                    const float* __restrict__ x,
                                    float* __restrict__ agg,
                                    float* __restrict__ cnt,
                                    int E, int dlog) {
  long idx = (long)blockIdx.x * blockDim.x + threadIdx.x;
  if (idx >= ((long)E << dlog)) return;
  int e = (int)(idx >> dlog), c = (int)(idx & ((1 << dlog) - 1));
  int s = src[e], d = dst[e];
  atomicAdd(&agg[((size_t)d << dlog) + c], x[((size_t)s << dlog) + c]);
  if (c == 0) atomicAdd(&cnt[d], 1.0f);
}
__global__ void sage_mean_kernel(float* __restrict__ agg,
                                 const float* __restrict__ cnt,
                                 int N, int dlog) {
  long idx = (long)blockIdx.x * blockDim.x + threadIdx.x;
  if (idx >= ((long)N << dlog)) return;
  agg[idx] /= fmaxf(cnt[idx >> dlog], 1.0f);
}

// BatchNorm (over batch dim, biased var, as in reference) + ReLU
__global__ void bn_stats_kernel(const float* __restrict__ x, int N,
                                float* __restrict__ mu, float* __restrict__ rstd) {
  __shared__ float s1[256], s2[256];
  int c = blockIdx.x, t = threadIdx.x;
  float a = 0.f, b = 0.f;
  for (int r = t; r < N; r += 256) {
    float v = x[(size_t)r * 64 + c];
    a += v; b += v * v;
  }
  s1[t] = a; s2[t] = b; __syncthreads();
  for (int o = 128; o > 0; o >>= 1) {
    if (t < o) { s1[t] += s1[t + o]; s2[t] += s2[t + o]; }
    __syncthreads();
  }
  if (t == 0) {
    float m = s1[0] / N;
    float var = s2[0] / N - m * m;
    mu[c] = m; rstd[c] = rsqrtf(var + 1e-5f);
  }
}
__global__ void bn_apply_kernel(const float* __restrict__ x,
                                const float* __restrict__ mu,
                                const float* __restrict__ rstd,
                                const float* __restrict__ g,
                                const float* __restrict__ b,
                                float* __restrict__ y, long n) {
  long i = (long)blockIdx.x * blockDim.x + threadIdx.x;
  if (i >= n) return;
  int c = (int)(i & 63);
  float v = (x[i] - mu[c]) * rstd[c] * g[c] + b[c];
  y[i] = fmaxf(v, 0.f);
}

// =====================================================================
// host orchestration
// =====================================================================
static inline int cdiv(long a, long b) { return (int)((a + b - 1) / b); }

extern "C" void kernel_launch(void* const* d_in, const int* in_sizes, int n_in,
                              void* d_out, int out_size, void* d_ws, size_t ws_size,
                              hipStream_t stream) {
  const float* xu0 = (const float*)d_in[0];
  const float* xr0 = (const float*)d_in[1];
  const float* xc0 = (const float*)d_in[2];
  const int* ei_ur = (const int*)d_in[3];
  const int* ei_ru = (const int*)d_in[4];
  const int* ei_rc = (const int*)d_in[5];
  const int* ei_cr = (const int*)d_in[6];
  const float* ea_ur = (const float*)d_in[7];
  const float* ea_ru = (const float*)d_in[8];
  // params flattened in dict insertion order; per layer (26 tensors):
  // ur: Ws Wd We att_s att_d att_e b | ru: same | rc: Wl bl Wr | cr: Wl bl Wr
  // bn_user g,b | bn_rest g,b | bn_cat g,b
  auto P = [&](int i) -> const float* { return (const float*)d_in[9 + i]; };

  // -------- workspace arenas (floats) --------
  float* ws = (float*)d_ws;
  size_t off = 0;
  auto alloc = [&](size_t nflt) { float* p = ws + off; off += nflt; return p; };
  float*    XU_N  = alloc((size_t)NU * 64);
  float*    XR_N  = alloc((size_t)NR * 64);
  float*    XC_N  = alloc((size_t)NC * 64);
  float*    HS    = alloc((size_t)NU * 64);   // also reused as SAGE agg
  float*    HD    = alloc((size_t)NU * 64);
  float*    OUTB  = alloc((size_t)NU * 64);   // shared pre-BN accumulator
  float*    AS    = alloc(NU);
  float*    AD    = alloc(NU);
  float*    ESUM  = alloc((size_t)NU * 16);
  float*    ECNT  = alloc(NU);                // also SAGE counts
  float*    LOGIT = alloc((size_t)E_REV + NU);
  unsigned* MKEY  = (unsigned*)alloc(NU);
  float*    SSUM  = alloc(NU);
  float*    WEATT = alloc(16);
  float*    MUB   = alloc(64);
  float*    RSTD  = alloc(64);
  if (ws_size < off * sizeof(float)) return;  // insufficient scratch

  auto fillf = [&](float* p, float v, long n) {
    fill_f32_kernel<<<cdiv(n, 256), 256, 0, stream>>>(p, v, n);
  };
  auto gemm = [&](const float* A, const float* W, float* C, int M, int K, int acc) {
    int g = cdiv(M, 32);
    if (K == 32) {
      if (acc) gemm_wmma_kernel<32, true ><<<g, 128, 0, stream>>>(A, W, C, M);
      else     gemm_wmma_kernel<32, false><<<g, 128, 0, stream>>>(A, W, C, M);
    } else {
      if (acc) gemm_wmma_kernel<64, true ><<<g, 128, 0, stream>>>(A, W, C, M);
      else     gemm_wmma_kernel<64, false><<<g, 128, 0, stream>>>(A, W, C, M);
    }
  };
  auto run_bn = [&](const float* x, int N, const float* g, const float* b, float* y) {
    bn_stats_kernel<<<64, 256, 0, stream>>>(x, N, MUB, RSTD);
    bn_apply_kernel<<<cdiv((long)N * 64, 256), 256, 0, stream>>>(
        x, MUB, RSTD, g, b, y, (long)N * 64);
  };

  // GAT(src-type -> dst-type): writes OUTB[0 : Nd*64) (bias skipped; BN-invariant)
  auto run_gat = [&](const float* xs, int Ns, int Ks, const float* xd, int Nd, int Kd,
                     const int* ei, const float* ea, int E,
                     const float* Wsm, const float* Wdm, const float* Wem,
                     const float* atts, const float* attd, const float* atte) {
    const int* src = ei;
    const int* dst = ei + E;
    int n_loop = (Ns < Nd) ? Ns : Nd;
    gemm(xs, Wsm, HS, Ns, Ks, 0);
    gemm(xd, Wdm, HD, Nd, Kd, 0);
    rowdot_kernel<<<cdiv(Ns, 4), 128, 0, stream>>>(HS, atts, AS, Ns);
    rowdot_kernel<<<cdiv(Nd, 4), 128, 0, stream>>>(HD, attd, AD, Nd);
    weatt_kernel<<<1, 16, 0, stream>>>(Wem, atte, WEATT);
    fillf(ESUM, 0.f, (long)Nd * 16);
    fillf(ECNT, 0.f, Nd);
    ea_scatter_kernel<<<cdiv((long)E * 16, 256), 256, 0, stream>>>(dst, ea, ESUM, ECNT, E);
    fill_u32_kernel<<<cdiv(Nd, 256), 256, 0, stream>>>(MKEY, 0x007FFFFFu, Nd); // key(-inf)
    gat_pass1_kernel<<<cdiv(E + n_loop, 256), 256, 0, stream>>>(
        src, dst, ea, ESUM, ECNT, AS, AD, WEATT, LOGIT, MKEY, E, n_loop);
    fillf(SSUM, 0.f, Nd);
    gat_pass2_kernel<<<cdiv(E + n_loop, 256), 256, 0, stream>>>(
        dst, LOGIT, MKEY, SSUM, E, n_loop);
    fillf(OUTB, 0.f, (long)Nd * 64);
    gat_pass3_kernel<<<cdiv((long)(E + n_loop) * 64, 256), 256, 0, stream>>>(
        src, dst, LOGIT, SSUM, HS, OUTB, E, n_loop);
  };

  // SAGE: out[0:Nd*64) += mean_agg(xs)@Wl + xd@Wr  (bl skipped; BN-invariant)
  auto run_sage = [&](const float* xs, int kslog, const float* xd, int Nd, int kdlog,
                      const int* ei, int E, const float* Wl, const float* Wr,
                      float* out) {
    const int* src = ei;
    const int* dst = ei + E;
    fillf(HS, 0.f, (long)Nd << kslog);     // HS arena reused as agg
    fillf(ECNT, 0.f, Nd);
    sage_scatter_kernel<<<cdiv((long)E << kslog, 256), 256, 0, stream>>>(
        src, dst, xs, HS, ECNT, E, kslog);
    sage_mean_kernel<<<cdiv((long)Nd << kslog, 256), 256, 0, stream>>>(HS, ECNT, Nd, kslog);
    gemm(HS, Wl, out, Nd, 1 << kslog, 1);
    gemm(xd, Wr, out, Nd, 1 << kdlog, 1);
  };

  auto layer = [&](const float* xu, const float* xr, const float* xc, int dclog,
                   int lb, float* yu, float* yr, float* yc) {
    // ---- restaurants: GAT(user->rest) + SAGE(cat->rest), then BN ----
    run_gat(xu, NU, 64, xr, NR, 64, ei_ur, ea_ur, E_REV,
            P(lb + 0), P(lb + 1), P(lb + 2), P(lb + 3), P(lb + 4), P(lb + 5));
    run_sage(xc, dclog, xr, NR, 6, ei_cr, E_BEL, P(lb + 17), P(lb + 19), OUTB);
    run_bn(OUTB, NR, P(lb + 22), P(lb + 23), yr);
    // ---- users: GAT(rest->user), then BN ----
    run_gat(xr, NR, 64, xu, NU, 64, ei_ru, ea_ru, E_REV,
            P(lb + 7), P(lb + 8), P(lb + 9), P(lb + 10), P(lb + 11), P(lb + 12));
    run_bn(OUTB, NU, P(lb + 20), P(lb + 21), yu);
    // ---- categories: SAGE(rest->cat), then BN ----
    fillf(OUTB, 0.f, (long)NC * 64);
    run_sage(xr, 6, xc, NC, dclog, ei_rc, E_BEL, P(lb + 14), P(lb + 16), OUTB);
    run_bn(OUTB, NC, P(lb + 24), P(lb + 25), yc);
  };

  float* out_u = (float*)d_out;
  float* out_r = out_u + (size_t)NU * 64;
  float* out_c = out_r + (size_t)NR * 64;

  layer(xu0, xr0, xc0, /*dc=32*/ 5, 0,  XU_N, XR_N, XC_N);       // layer 1
  layer(XU_N, XR_N, XC_N, /*dc=64*/ 6, 26, out_u, out_r, out_c); // layer 2
}
